// HierarchicalAttentionEncoderLayer_32865089749465
// MI455X (gfx1250) — compile-verified, matrix operand order checked by reading, not run on hardware
//
#include <hip/hip_runtime.h>
#include <hip/hip_bf16.h>

// ---------------------------------------------------------------------------
// HierarchicalAttentionEncoderLayer for MI455X (gfx1250, wave32, WMMA).
// All GEMMs: bf16 WMMA (v_wmma_f32_16x16x32_bf16), fp32 accumulate.
// Block tile 128x64x64, 8 waves in a 4x2 grid, 32x32 per wave (4 accums,
// 8 WMMA per k-stage).  A and B both staged [row][k] in LDS so every
// fragment load is two contiguous ds_load_b128 per lane.
// ---------------------------------------------------------------------------

typedef __attribute__((ext_vector_type(16))) __bf16 v16bf;
typedef __attribute__((ext_vector_type(8)))  float  v8f;

constexpr int kD  = 768;
constexpr int kH  = 12;
constexpr int kDH = 64;
constexpr int kFF = 3072;
constexpr int kB  = 8;
constexpr int kS  = 512;
constexpr int kM  = kB * kS;            // 4096 rows of activations
constexpr int kLdsPitch = 72;           // 64 k + 8 pad (bank-conflict dodge)

// ---------------- WMMA fragment load (16-bit 16x32 layout, ISA 7.12.2) ------
// Lane L(0-15) holds row r0+L; kh = 8*(L>=16).  Per lane the 16 bf16 are two
// contiguous 8-element (16B) runs -> two ds_load_b128.
__device__ __forceinline__ v16bf load_frag(const __bf16* s, int r0, int koff, int lane) {
  int r  = r0 + (lane & 15);
  int kh = (lane >> 4) << 3;
  const __bf16* p = s + r * kLdsPitch + koff + kh;
  v16bf a;
#pragma unroll
  for (int i = 0; i < 8; ++i) a[i]     = p[i];
#pragma unroll
  for (int i = 0; i < 8; ++i) a[8 + i] = p[16 + i];
  return a;
}

// ---------------- Generic batched GEMM: C = act(alpha*A@B + bias + R) -------
__global__ __launch_bounds__(256)
void gemm_wmma_bf16(const float* __restrict__ Ab, int lda, long As1, long As2,
                    const float* __restrict__ Bb, int ldb, long Bs1, long Bs2, int BT,
                    const float* __restrict__ bias, const float* __restrict__ Rb,
                    float* __restrict__ Cb, int ldc, long Cs1, long Cs2,
                    int Ksz, float alpha, int relu, int Hdiv)
{
  __shared__ __bf16 sA[128 * kLdsPitch];  // [m][k]
  __shared__ __bf16 sB[64 * kLdsPitch];   // [n][k]  (column-major stage of B)

  const int tid  = threadIdx.x;
  const int lane = tid & 31;
  const int wave = tid >> 5;
  const int wm   = wave >> 1;        // 0..3  -> 32-row band
  const int wn   = wave & 1;         // 0..1  -> 32-col band

  const int z  = blockIdx.z;
  const int zb = z / Hdiv;
  const int zh = z - zb * Hdiv;
  const float* A  = Ab + (long)zb * As1 + (long)zh * As2;
  const float* Bm = Bb + (long)zb * Bs1 + (long)zh * Bs2;
  float*       C  = Cb + (long)zb * Cs1 + (long)zh * Cs2;
  const float* R  = Rb ? (Rb + (long)zb * Cs1 + (long)zh * Cs2) : nullptr;

  const int m0 = blockIdx.y * 128;
  const int n0 = blockIdx.x * 64;

  v8f acc00 = {};   // (m 0..15,  n 0..15)
  v8f acc01 = {};   // (m 0..15,  n 16..31)
  v8f acc10 = {};   // (m 16..31, n 0..15)
  v8f acc11 = {};   // (m 16..31, n 16..31)

  // Staging maps: A tile 128x64 (32 elem/thread), B tile 64x64 (16 elem/thread)
  const int asr = tid >> 1;            // 0..127
  const int asc = (tid & 1) * 32;      // 0 or 32
  const int bsr = tid >> 2;            // 0..63
  const int bsc = (tid & 3) * 16;      // 0,16,32,48

  for (int k0 = 0; k0 < Ksz; k0 += 64) {
    {   // A tile: contiguous global read, contiguous LDS write
      const float* src = A + (long)(m0 + asr) * lda + (k0 + asc);
      __bf16* dst = sA + asr * kLdsPitch + asc;
#pragma unroll
      for (int j = 0; j < 32; ++j) dst[j] = (__bf16)src[j];
      if (k0 + 64 < Ksz) __builtin_prefetch(src + 64, 0, 0);
    }
    if (BT) {   // B source [N][K]: contiguous read, contiguous LDS write
      const float* src = Bm + (long)(n0 + bsr) * ldb + (k0 + bsc);
      __bf16* dst = sB + bsr * kLdsPitch + bsc;
#pragma unroll
      for (int j = 0; j < 16; ++j) dst[j] = (__bf16)src[j];
      if (k0 + 64 < Ksz) __builtin_prefetch(src + 64, 0, 0);
    } else {    // B source [K][N]: contiguous read along n, strided LDS write
      const float* src = Bm + (long)(k0 + bsr) * ldb + (n0 + bsc);
#pragma unroll
      for (int j = 0; j < 16; ++j) sB[(bsc + j) * kLdsPitch + bsr] = (__bf16)src[j];
      if (k0 + 64 < Ksz) __builtin_prefetch(Bm + (long)(k0 + 64 + bsr) * ldb + (n0 + bsc), 0, 0);
    }
    __syncthreads();

#pragma unroll
    for (int kk = 0; kk < 64; kk += 32) {
      v16bf a0 = load_frag(sA, wm * 32,      kk, lane);
      v16bf a1 = load_frag(sA, wm * 32 + 16, kk, lane);
      v16bf b0 = load_frag(sB, wn * 32,      kk, lane);
      v16bf b1 = load_frag(sB, wn * 32 + 16, kk, lane);
      acc00 = __builtin_amdgcn_wmma_f32_16x16x32_bf16(false, a0, false, b0,
                                                      (short)0, acc00, false, false);
      acc01 = __builtin_amdgcn_wmma_f32_16x16x32_bf16(false, a0, false, b1,
                                                      (short)0, acc01, false, false);
      acc10 = __builtin_amdgcn_wmma_f32_16x16x32_bf16(false, a1, false, b0,
                                                      (short)0, acc10, false, false);
      acc11 = __builtin_amdgcn_wmma_f32_16x16x32_bf16(false, a1, false, b1,
                                                      (short)0, acc11, false, false);
    }
    __syncthreads();
  }

  // Epilogue.  C/D layout (ISA 7.12.2): lane L holds column n=L&15,
  // rows mb..mb+7 with mb = 8*(L>=16) across the 8 accumulator VGPRs.
  const int cn = lane & 15;
  const int mb = (lane >> 4) * 8;
  const int gn0 = n0 + wn * 32 + cn;
  const int gn1 = gn0 + 16;
  const float bias_v0 = bias ? bias[gn0] : 0.f;
  const float bias_v1 = bias ? bias[gn1] : 0.f;
#pragma unroll
  for (int i = 0; i < 8; ++i) {
    const int row0 = m0 + wm * 32 + mb + i;
    const int row1 = row0 + 16;

    float v00 = alpha * acc00[i] + bias_v0;
    float v01 = alpha * acc01[i] + bias_v1;
    float v10 = alpha * acc10[i] + bias_v0;
    float v11 = alpha * acc11[i] + bias_v1;
    if (R) {
      v00 += R[(long)row0 * ldc + gn0];
      v01 += R[(long)row0 * ldc + gn1];
      v10 += R[(long)row1 * ldc + gn0];
      v11 += R[(long)row1 * ldc + gn1];
    }
    if (relu) {
      v00 = fmaxf(v00, 0.f); v01 = fmaxf(v01, 0.f);
      v10 = fmaxf(v10, 0.f); v11 = fmaxf(v11, 0.f);
    }
    C[(long)row0 * ldc + gn0] = v00;
    C[(long)row0 * ldc + gn1] = v01;
    C[(long)row1 * ldc + gn0] = v10;
    C[(long)row1 * ldc + gn1] = v11;
  }
}

// ---------------- LayerNorm over D=768, one row per 256-thread block --------
__global__ __launch_bounds__(256)
void layernorm_k(const float* __restrict__ x, const float* __restrict__ g,
                 const float* __restrict__ b, float* __restrict__ y)
{
  const int tid = threadIdx.x;
  const float* xr = x + (long)blockIdx.x * kD;
  float* yr = y + (long)blockIdx.x * kD;

  float s = 0.f, ss = 0.f;
  for (int i = tid; i < kD; i += 256) { float v = xr[i]; s += v; ss += v * v; }

  __shared__ float r1[256], r2[256];
  r1[tid] = s; r2[tid] = ss;
  __syncthreads();
  for (int off = 128; off > 0; off >>= 1) {
    if (tid < off) { r1[tid] += r1[tid + off]; r2[tid] += r2[tid + off]; }
    __syncthreads();
  }
  const float mean = r1[0] / kD;
  const float var  = r2[0] / kD - mean * mean;
  const float inv  = rsqrtf(var + 1e-5f);
  for (int i = tid; i < kD; i += 256)
    yr[i] = (xr[i] - mean) * inv * g[i] + b[i];
}

// ---------------- Neighbor affinity: a[b,t] from qn/kn dot products ---------
__global__ __launch_bounds__(256)
void affinity_k(const float* __restrict__ qn, const float* __restrict__ kn,
                const float* __restrict__ prev_a, const int* __restrict__ layer_idx,
                float* __restrict__ a_out)
{
  const int idx = blockIdx.x;                 // 0 .. B*(S-1)-1
  const int b = idx / (kS - 1);
  const int t = idx - b * (kS - 1);
  const float* q0 = qn + ((long)b * kS + t) * kD;
  const float* k0 = kn + ((long)b * kS + t) * kD;
  const float* q1 = q0 + kD;
  const float* k1 = k0 + kD;

  const int tid = threadIdx.x;
  float sf = 0.f, sb = 0.f;
  for (int i = tid; i < kD; i += 256) { sf += q0[i] * k1[i]; sb += q1[i] * k0[i]; }

  __shared__ float rf[256], rb[256];
  rf[tid] = sf; rb[tid] = sb;
  __syncthreads();
  for (int off = 128; off > 0; off >>= 1) {
    if (tid < off) { rf[tid] += rf[tid + off]; rb[tid] += rb[tid + off]; }
    __syncthreads();
  }
  if (tid == 0) {
    float f = rf[0] * (1.f / 64.f);
    float w = rb[0] * (1.f / 64.f);
    float ah = 0.5f * (1.f / (1.f + expf(-f)) + 1.f / (1.f + expf(-w)));
    float r = (*layer_idx == 0) ? ah : prev_a[idx] + (1.f - prev_a[idx]) * ah;
    a_out[idx] = r;
  }
}

// ---------------- Per-batch exclusive log-cumsum L[k] = sum_{t<k} log a[t] --
__global__ __launch_bounds__(512)
void cumsum_k(const float* __restrict__ a, float* __restrict__ Lb)
{
  __shared__ float sc[512];
  const int b = blockIdx.x, t = threadIdx.x;
  sc[t] = (t < kS - 1) ? logf(a[(long)b * (kS - 1) + t]) : 0.f;
  __syncthreads();
  for (int off = 1; off < 512; off <<= 1) {        // Hillis-Steele inclusive
    float v = (t >= off) ? sc[t - off] : 0.f;
    __syncthreads();
    sc[t] += v;
    __syncthreads();
  }
  Lb[(long)b * kS + t] = (t == 0) ? 0.f : sc[t - 1];
}

// ---------------- Row softmax fused with hierarchical mask C ----------------
__global__ __launch_bounds__(256)
void softmax_c_k(float* __restrict__ attn, const unsigned char* __restrict__ mask,
                 const float* __restrict__ Lb)
{
  const long row = blockIdx.x;                // 0 .. B*H*S-1
  const int  i   = (int)(row & (kS - 1));
  const int  bh  = (int)(row / kS);
  const int  b   = bh / kH;
  float* p = attn + row * (long)kS;
  const unsigned char* mk = mask + (long)b * kS;
  const float* L = Lb + (long)b * kS;

  const int tid = threadIdx.x;
  const int j0 = tid, j1 = tid + 256;
  float v0 = mk[j0] ? p[j0] : -1e9f;
  float v1 = mk[j1] ? p[j1] : -1e9f;

  __shared__ float red[256];
  red[tid] = fmaxf(v0, v1);
  __syncthreads();
  for (int off = 128; off > 0; off >>= 1) {
    if (tid < off) red[tid] = fmaxf(red[tid], red[tid + off]);
    __syncthreads();
  }
  const float rowmax = red[0];
  __syncthreads();

  const float e0 = expf(v0 - rowmax);
  const float e1 = expf(v1 - rowmax);
  red[tid] = e0 + e1;
  __syncthreads();
  for (int off = 128; off > 0; off >>= 1) {
    if (tid < off) red[tid] += red[tid + off];
    __syncthreads();
  }
  const float inv = 1.f / red[0];

  const float Li = L[i];
  // C[i,j] = exp(L[max(i,j)] - L[min(i,j)])
  float c0 = expf((j0 > i) ? (L[j0] - Li) : (Li - L[j0]));
  float c1 = expf((j1 > i) ? (L[j1] - Li) : (Li - L[j1]));
  p[j0] = e0 * inv * c0;
  p[j1] = e1 * inv * c1;
}

// ---------------------------------------------------------------------------
static inline void launch_gemm(hipStream_t st,
    const float* A, int lda, long As1, long As2,
    const float* Bm, int ldb, long Bs1, long Bs2, int BT,
    const float* bias, const float* R,
    float* C, int ldc, long Cs1, long Cs2,
    int M, int N, int K, float alpha, int relu, int Hdiv, int batches)
{
  dim3 grid(N / 64, M / 128, batches);
  gemm_wmma_bf16<<<grid, 256, 0, st>>>(A, lda, As1, As2, Bm, ldb, Bs1, Bs2, BT,
                                       bias, R, C, ldc, Cs1, Cs2, K, alpha, relu, Hdiv);
}

extern "C" void kernel_launch(void* const* d_in, const int* in_sizes, int n_in,
                              void* d_out, int out_size, void* d_ws, size_t ws_size,
                              hipStream_t stream)
{
  const float* x      = (const float*)d_in[0];
  const unsigned char* mask = (const unsigned char*)d_in[1];   // jnp bool -> 1 byte
  const float* prev_a = (const float*)d_in[2];
  const float* Wqn = (const float*)d_in[3];  const float* bqn = (const float*)d_in[4];
  const float* Wkn = (const float*)d_in[5];  const float* bkn = (const float*)d_in[6];
  const float* Wq  = (const float*)d_in[7];  const float* bq  = (const float*)d_in[8];
  const float* Wk  = (const float*)d_in[9];  const float* bk  = (const float*)d_in[10];
  const float* Wv  = (const float*)d_in[11]; const float* bv  = (const float*)d_in[12];
  const float* Wo  = (const float*)d_in[13]; const float* bo  = (const float*)d_in[14];
  const float* W1  = (const float*)d_in[15]; const float* b1  = (const float*)d_in[16];
  const float* W2  = (const float*)d_in[17]; const float* b2  = (const float*)d_in[18];
  const float* g1  = (const float*)d_in[19]; const float* be1 = (const float*)d_in[20];
  const float* g2  = (const float*)d_in[21]; const float* be2 = (const float*)d_in[22];
  const int*   layer_idx = (const int*)d_in[23];

  const long MD = (long)kM * kD;                 // 3,145,728
  float* out   = (float*)d_out;                  // [B,S,D]
  float* a_out = out + MD;                       // [B,S-1]
  float* attn  = a_out + (long)kB * (kS - 1);    // [B,H,S,S]

  float* ws  = (float*)d_ws;
  float* XN  = ws;                  // ln1 output, later reused as ln2 output
  float* Qb  = XN + MD;             // qn, then q, then ctx ([B,S,D])
  float* Kb  = Qb + MD;             // kn, then k, then x1
  float* Vb  = Kb + MD;             // v
  float* H1  = Vb + MD;             // FFN hidden [4096,3072]
  float* Lb  = H1 + (long)kM * kFF; // [B,S] log prefix sums
  float* CTX = Qb;                  // alias (q dead after scores)
  float* X1  = Kb;                  // alias (k dead after scores)
  float* XN2 = XN;                  // alias (xn dead after q/k/v)

  const long SD = (long)kS * kD;    // per-batch activation stride
  const long SS = (long)kS * kS;    // per-(b,h) score stride

  // 1) neighbor projections on raw x, then affinity + log-cumsum
  launch_gemm(stream, x, kD, 0, 0, Wqn, kD, 0, 0, 0, bqn, nullptr,
              Qb, kD, 0, 0, kM, kD, kD, 1.f, 0, 1, 1);
  launch_gemm(stream, x, kD, 0, 0, Wkn, kD, 0, 0, 0, bkn, nullptr,
              Kb, kD, 0, 0, kM, kD, kD, 1.f, 0, 1, 1);
  affinity_k<<<dim3(kB * (kS - 1)), 256, 0, stream>>>(Qb, Kb, prev_a, layer_idx, a_out);
  cumsum_k<<<dim3(kB), 512, 0, stream>>>(a_out, Lb);

  // 2) ln1 + q/k/v projections
  layernorm_k<<<dim3(kM), 256, 0, stream>>>(x, g1, be1, XN);
  launch_gemm(stream, XN, kD, 0, 0, Wq, kD, 0, 0, 0, bq, nullptr,
              Qb, kD, 0, 0, kM, kD, kD, 1.f, 0, 1, 1);
  launch_gemm(stream, XN, kD, 0, 0, Wk, kD, 0, 0, 0, bk, nullptr,
              Kb, kD, 0, 0, kM, kD, kD, 1.f, 0, 1, 1);
  launch_gemm(stream, XN, kD, 0, 0, Wv, kD, 0, 0, 0, bv, nullptr,
              Vb, kD, 0, 0, kM, kD, kD, 1.f, 0, 1, 1);

  // 3) scores = Q K^T / sqrt(64) into attn region (batched over b,h; BT=1)
  launch_gemm(stream, Qb, kD, SD, kDH, Kb, kD, SD, kDH, 1, nullptr, nullptr,
              attn, kS, (long)kH * SS, SS, kS, kS, kDH, 0.125f, 0, kH, kB * kH);

  // 4) softmax + hierarchical mask C (writes final attn_h output in place)
  softmax_c_k<<<dim3(kB * kH * kS), 256, 0, stream>>>(attn, mask, Lb);

  // 5) ctx = attn_h @ V, stored directly in [B,S,D] interleaved-head layout
  launch_gemm(stream, attn, kS, (long)kH * SS, SS, Vb, kD, SD, kDH, 0,
              nullptr, nullptr, CTX, kD, SD, kDH, kS, kDH, kS, 1.f, 0, kH, kB * kH);

  // 6) x1 = x + ctx @ Wo + bo
  launch_gemm(stream, CTX, kD, 0, 0, Wo, kD, 0, 0, 0, bo, x,
              X1, kD, 0, 0, kM, kD, kD, 1.f, 0, 1, 1);

  // 7) ln2 + FFN with fused ReLU, final residual into d_out
  layernorm_k<<<dim3(kM), 256, 0, stream>>>(X1, g2, be2, XN2);
  launch_gemm(stream, XN2, kD, 0, 0, W1, kFF, 0, 0, 0, b1, nullptr,
              H1, kFF, 0, 0, kM, kFF, kD, 1.f, 1, 1, 1);
  launch_gemm(stream, H1, kFF, 0, 0, W2, kD, 0, 0, 0, b2, X1,
              out, kD, 0, 0, kM, kD, kFF, 1.f, 0, 1, 1);
}
